// GGNN_34402688041271
// MI455X (gfx1250) — matrix operand order repaired and checked by reading, not compiled, not run on hardware
//
#include <hip/hip_runtime.h>
#include <hip/hip_bf16.h>

#define B_  16
#define N_  512
#define U_  128
#define E_  8
#define M_  (B_ * N_)   // 8192 rows (b,n)
#define EU_ (E_ * U_)   // 1024
#define K3_ 384         // 3*U

typedef __attribute__((ext_vector_type(16))) __bf16 v16bf;
typedef __attribute__((ext_vector_type(8)))  float  v8f;

__device__ inline v8f wmma_bf16(v16bf a, v16bf b, v8f c) {
    // D = A(16x32 bf16) * B(32x16 bf16) + C(f32)
    return __builtin_amdgcn_wmma_f32_16x16x32_bf16(
        false, a, false, b, (short)0, c, false, false);
}

// ---- WMMA operand loaders (layouts per CDNA5 ISA 7.12.2) ----------------
// A 16x32 bf16 from row-major [M][K]: two contiguous 16B chunks per lane.
__device__ inline v16bf ldA_bf16(const __bf16* __restrict__ src, int ld,
                                 int m0, int k0, int lane) {
    int row = m0 + (lane & 15);
    int kh  = (lane >> 4) << 3;
    const __bf16* p = src + row * ld + k0 + kh;
    v16bf a;
#pragma unroll
    for (int j = 0; j < 8; ++j) a[j]     = p[j];
#pragma unroll
    for (int j = 0; j < 8; ++j) a[8 + j] = p[16 + j];
    return a;
}

// A 16x32 from row-major f32 (only used for the small `state` operand).
__device__ inline v16bf ldA_f32(const float* __restrict__ src, int ld,
                                int m0, int k0, int lane) {
    int row = m0 + (lane & 15);
    int kh  = (lane >> 4) << 3;
    const float* p = src + row * ld + k0 + kh;
    v16bf a;
#pragma unroll
    for (int j = 0; j < 8; ++j) a[j]     = (__bf16)p[j];
#pragma unroll
    for (int j = 0; j < 8; ++j) a[8 + j] = (__bf16)p[16 + j];
    return a;
}

// B 32x16 bf16 from K-contiguous ("transposed") [N][K] storage:
// two contiguous 16B chunks per lane.
__device__ inline v16bf ldB_bf16T(const __bf16* __restrict__ src, int ldK,
                                  int k0, int n0, int lane) {
    int n  = n0 + (lane & 15);
    int kb = k0 + ((lane >> 4) << 3);
    const __bf16* p = src + n * ldK + kb;
    v16bf b;
#pragma unroll
    for (int j = 0; j < 8; ++j) b[j]     = p[j];
#pragma unroll
    for (int j = 0; j < 8; ++j) b[8 + j] = p[16 + j];
    return b;
}

// ======================= one-time prep kernels ===========================
// wadj_bf = bf16(adj * sigmoid(alpha[e]));  4 elements / thread
__global__ __launch_bounds__(256) void k_prep_adj(
    const float* __restrict__ adj, const float* __restrict__ alpha,
    __bf16* __restrict__ out) {
    int base = (blockIdx.x * 256 + threadIdx.x) * 4;
    int e = (base >> 18) & 7;                 // N*N = 2^18, layout [b][e][n][m]
    float s = 1.0f / (1.0f + __expf(-alpha[e]));
    float4 v = *(const float4*)(adj + base);
    out[base + 0] = (__bf16)(v.x * s);
    out[base + 1] = (__bf16)(v.y * s);
    out[base + 2] = (__bf16)(v.z * s);
    out[base + 3] = (__bf16)(v.w * s);
}

// wT[e][u][d] = bf16(w[e][d][u])
__global__ __launch_bounds__(256) void k_prep_w(
    const float* __restrict__ w, __bf16* __restrict__ wT) {
    int i = blockIdx.x * 256 + threadIdx.x;   // E*U*U = 131072
    int e = i >> 14, r = i & 16383, u = r >> 7, d = r & 127;
    wT[i] = (__bf16)w[(e << 14) + (d << 7) + u];
}

// gkT[n][k] = bf16(gk[k][n]),  n<384, k<1024
__global__ __launch_bounds__(256) void k_prep_gk(
    const float* __restrict__ gk, __bf16* __restrict__ gkT) {
    int i = blockIdx.x * 256 + threadIdx.x;   // 393216
    int n = i >> 10, k = i & 1023;
    gkT[i] = (__bf16)gk[k * K3_ + n];
}

// recT[n][k] = bf16(rec[k][n]),  n<384, k<128
__global__ __launch_bounds__(256) void k_prep_rec(
    const float* __restrict__ rec, __bf16* __restrict__ recT) {
    int i = blockIdx.x * 256 + threadIdx.x;   // 49152
    int n = i >> 7, k = i & 127;
    recT[i] = (__bf16)rec[k * K3_ + n];
}

// ---- Stage 1: hT[b,e,u,m] = (state[b,:,:] @ w[e] + bias[e])^T (bf16) -----
// LDS transpose epilogue -> K-contiguous B operand for stage 2.
__global__ __launch_bounds__(256) void k_edge_transform(
    const float* __restrict__ state,  // [M_, U_]
    const __bf16* __restrict__ wT,    // [E_, U_, U_] (u-major)
    const float* __restrict__ bias,   // [E_, U_]
    __bf16* __restrict__ hT)          // [B_, E_, U_, N_]
{
    __shared__ __bf16 sbuf[8 * 64 * 24];     // per-wave 64 cols x 24-stride
    int wv   = threadIdx.x >> 5;
    int lane = threadIdx.x & 31;
    int wid  = blockIdx.x * 8 + wv;
    int e    = wid >> 10;                    // 1024 waves per edge type
    int rem  = wid & 1023;
    int m0   = (rem >> 1) << 4;              // 512 m-tiles (global row)
    int n0   = (rem & 1) << 6;               // 2 groups of 64 cols
    const __bf16* we = wT + e * U_ * U_;

    v8f acc[4] = {};
    for (int k0 = 0; k0 < U_; k0 += 32) {
        v16bf a = ldA_f32(state, U_, m0, k0, lane);
        v16bf bm[4];
#pragma unroll
        for (int t = 0; t < 4; ++t)
            bm[t] = ldB_bf16T(we, U_, k0, n0 + (t << 4), lane);
#pragma unroll
        for (int t = 0; t < 4; ++t)
            acc[t] = wmma_bf16(a, bm[t], acc[t]);
    }
    // stage C tile (16 rows x 64 cols) into LDS, transposed access pattern
    __bf16* s = sbuf + wv * 64 * 24;
    const float* be = bias + e * U_;
    int rowl = (lane >> 4) << 3;             // 0 or 8
#pragma unroll
    for (int t = 0; t < 4; ++t) {
        int col = (t << 4) + (lane & 15);    // local col 0..63
        float bv = be[n0 + col];
#pragma unroll
        for (int r = 0; r < 8; ++r)
            s[col * 24 + rowl + r] = (__bf16)(acc[t][r] + bv);
    }
    __syncthreads();
    // write back: each lane emits two u-rows of 16 nodes (32B contiguous)
    int bb   = m0 >> 9;
    int node = m0 & 511;
#pragma unroll
    for (int q = 0; q < 2; ++q) {
        int u = (lane << 1) + q;             // local col 0..63
        const __bf16* sp = s + u * 24;
        __bf16* dp = hT + ((size_t)((bb * E_ + e) * U_ + n0 + u)) * N_ + node;
#pragma unroll
        for (int j = 0; j < 16; ++j) dp[j] = sp[j];
    }
}

// ---- Stage 2: neigh[b,n,e,:] = wadj_bf[b,e] @ h[b,e]  (32x64 per wave) ---
__global__ __launch_bounds__(256) void k_aggregate(
    const __bf16* __restrict__ wadj,  // [B_, E_, N_, N_] bf16, alpha-scaled
    const __bf16* __restrict__ hT,    // [B_, E_, U_, N_]
    __bf16* __restrict__ neigh)       // [B_, N_, E_, U_] == [M_, EU_]
{
    int wid  = blockIdx.x * 8 + (threadIdx.x >> 5);
    int lane = threadIdx.x & 31;
    int be   = wid >> 5;              // 32 waves per (b,e)
    int bb   = be >> 3;
    int e    = be & 7;
    int rem  = wid & 31;
    int m0   = (rem >> 1) << 5;       // 16 m-tiles of 32 rows
    int n0   = (rem & 1) << 6;        // 2 groups of 64 cols
    const __bf16* A  = wadj + (size_t)(bb * E_ + e) * N_ * N_;
    const __bf16* Bt = hT   + (size_t)(bb * E_ + e) * U_ * N_;

    v8f acc[2][4] = {};
    for (int k0 = 0; k0 < N_; k0 += 32) {
        if (k0 + 128 < N_)  // keep the adjacency stream ahead in cache
            __builtin_prefetch(A + (m0 + (lane & 15)) * N_ + k0 + 128, 0, 1);
        // issue all operand loads for this k-step first (single clause)
        v16bf a0 = ldA_bf16(A, N_, m0,      k0, lane);
        v16bf a1 = ldA_bf16(A, N_, m0 + 16, k0, lane);
        v16bf bm[4];
#pragma unroll
        for (int t = 0; t < 4; ++t)
            bm[t] = ldB_bf16T(Bt, N_, k0, n0 + (t << 4), lane);
#pragma unroll
        for (int t = 0; t < 4; ++t) {
            acc[0][t] = wmma_bf16(a0, bm[t], acc[0][t]);
            acc[1][t] = wmma_bf16(a1, bm[t], acc[1][t]);
        }
    }
    __bf16* dst = neigh + (size_t)bb * N_ * EU_ + e * U_;
#pragma unroll
    for (int h = 0; h < 2; ++h) {
        int nrow = m0 + (h << 4) + ((lane >> 4) << 3);
#pragma unroll
        for (int t = 0; t < 4; ++t) {
            int col = n0 + (t << 4) + (lane & 15);
#pragma unroll
            for (int r = 0; r < 8; ++r)
                dst[(nrow + r) * EU_ + col] = (__bf16)acc[h][t][r];
        }
    }
}

// ---- Stage 3a: tmp = gru_in @ K + bias  (+ h @ U_rec for z/r columns) ----
// 16x64 tile per wave; 384 cols = 6 groups; z/r boundary (256) = group 4.
__global__ __launch_bounds__(256) void k_gru_gemm(
    const __bf16* __restrict__ neigh, // [M_, EU_]
    const float* __restrict__ state,  // [M_, U_]
    const __bf16* __restrict__ gkT,   // [K3_, EU_]
    const __bf16* __restrict__ recT,  // [K3_, U_]
    const float* __restrict__ gb,     // [K3_]
    float* __restrict__ tmp)          // [M_, K3_]
{
    int wid  = blockIdx.x * 8 + (threadIdx.x >> 5);   // 3072 waves
    int lane = threadIdx.x & 31;
    int m0   = (wid / 6) << 4;
    int n0   = (wid % 6) << 6;

    v8f acc[4] = {};
    for (int k0 = 0; k0 < EU_; k0 += 32) {
        v16bf a = ldA_bf16(neigh, EU_, m0, k0, lane);
        v16bf bm[4];
#pragma unroll
        for (int t = 0; t < 4; ++t)
            bm[t] = ldB_bf16T(gkT, EU_, k0, n0 + (t << 4), lane);
#pragma unroll
        for (int t = 0; t < 4; ++t)
            acc[t] = wmma_bf16(a, bm[t], acc[t]);
    }
    if (n0 < 2 * U_) {                // z,r columns also get h @ rec
        for (int k0 = 0; k0 < U_; k0 += 32) {
            v16bf a = ldA_f32(state, U_, m0, k0, lane);
            v16bf bm[4];
#pragma unroll
            for (int t = 0; t < 4; ++t)
                bm[t] = ldB_bf16T(recT, U_, k0, n0 + (t << 4), lane);
#pragma unroll
            for (int t = 0; t < 4; ++t)
                acc[t] = wmma_bf16(a, bm[t], acc[t]);
        }
    }
    int row = m0 + ((lane >> 4) << 3);
#pragma unroll
    for (int t = 0; t < 4; ++t) {
        int col = n0 + (t << 4) + (lane & 15);
        float bv = gb[col];
#pragma unroll
        for (int r = 0; r < 8; ++r)
            tmp[(row + r) * K3_ + col] = acc[t][r] + bv;
    }
}

// ---- Stage 3b: gates z, r; materialize rh = bf16(r * h) ------------------
__global__ __launch_bounds__(256) void k_gru_gates(
    const float* __restrict__ tmp, const float* __restrict__ state,
    float* __restrict__ zbuf, __bf16* __restrict__ rh)
{
    int i   = blockIdx.x * 256 + threadIdx.x;
    int row = i >> 7, u = i & (U_ - 1);
    float xz = tmp[row * K3_ + u];
    float xr = tmp[row * K3_ + U_ + u];
    float h  = state[i];
    float zv = 1.0f / (1.0f + __expf(-xz));
    float rv = 1.0f / (1.0f + __expf(-xr));
    zbuf[i] = zv;
    rh[i]   = (__bf16)(rv * h);
}

// ---- Stage 3c: hh = tanh(xh + rh @ Uh); h' = z*h + (1-z)*hh --------------
__global__ __launch_bounds__(256) void k_gru_update(
    const float* __restrict__ tmp, const float* __restrict__ zbuf,
    const __bf16* __restrict__ rh, const __bf16* __restrict__ recT,
    const float* __restrict__ state, float* __restrict__ out)
{
    int wid  = blockIdx.x * 8 + (threadIdx.x >> 5);
    int lane = threadIdx.x & 31;
    int m0   = (wid >> 1) << 4;
    int n0   = (wid & 1) << 6;

    v8f acc[4] = {};
    for (int k0 = 0; k0 < U_; k0 += 32) {
        v16bf a = ldA_bf16(rh, U_, m0, k0, lane);
        v16bf bm[4];
#pragma unroll
        for (int t = 0; t < 4; ++t)
            bm[t] = ldB_bf16T(recT, U_, k0, 2 * U_ + n0 + (t << 4), lane);
#pragma unroll
        for (int t = 0; t < 4; ++t)
            acc[t] = wmma_bf16(a, bm[t], acc[t]);
    }
    int row0 = m0 + ((lane >> 4) << 3);
#pragma unroll
    for (int t = 0; t < 4; ++t) {
        int col = n0 + (t << 4) + (lane & 15);
#pragma unroll
        for (int r = 0; r < 8; ++r) {
            int row  = row0 + r;
            float xh = tmp[row * K3_ + 2 * U_ + col];
            float hh = tanhf(acc[t][r] + xh);
            float z  = zbuf[row * U_ + col];
            float h  = state[row * U_ + col];
            out[row * U_ + col] = z * h + (1.0f - z) * hh;
        }
    }
}

extern "C" void kernel_launch(void* const* d_in, const int* in_sizes, int n_in,
                              void* d_out, int out_size, void* d_ws, size_t ws_size,
                              hipStream_t stream) {
    (void)in_sizes; (void)n_in; (void)out_size; (void)ws_size;
    const float* init_state = (const float*)d_in[0];
    const float* adj        = (const float*)d_in[1];
    const float* alpha      = (const float*)d_in[2];
    const float* w          = (const float*)d_in[3];
    const float* bvec       = (const float*)d_in[4];
    const float* gk         = (const float*)d_in[5];
    const float* grk        = (const float*)d_in[6];
    const float* gb         = (const float*)d_in[7];

    char* ws = (char*)d_ws;
    size_t off = 0;
    auto alloc = [&](size_t bytes) {
        void* p = ws + off; off += (bytes + 255) & ~(size_t)255; return p;
    };
    __bf16* wadj  = (__bf16*)alloc((size_t)B_ * E_ * N_ * N_ * 2); // 67 MB
    __bf16* hT    = (__bf16*)alloc((size_t)B_ * E_ * U_ * N_ * 2); // 16 MB
    __bf16* neigh = (__bf16*)alloc((size_t)M_ * EU_ * 2);          // 16 MB
    float*  tmp   = (float*) alloc((size_t)M_ * K3_ * 4);          // 12.6 MB
    float*  zbuf  = (float*) alloc((size_t)M_ * U_ * 4);           // 4 MB
    __bf16* rhbuf = (__bf16*)alloc((size_t)M_ * U_ * 2);           // 2 MB
    float*  st    = (float*) alloc((size_t)M_ * U_ * 4);           // 4 MB
    __bf16* wT    = (__bf16*)alloc((size_t)E_ * U_ * U_ * 2);
    __bf16* gkT   = (__bf16*)alloc((size_t)K3_ * EU_ * 2);
    __bf16* recT  = (__bf16*)alloc((size_t)K3_ * U_ * 2);

    dim3 blk(256);  // 8 waves (wave32)
    // one-time (per launch) operand preparation: bf16 + K-contiguous layouts
    k_prep_adj<<<32768, blk, 0, stream>>>(adj, alpha, wadj);
    k_prep_w  <<<512,   blk, 0, stream>>>(w,   wT);
    k_prep_gk <<<1536,  blk, 0, stream>>>(gk,  gkT);
    k_prep_rec<<<192,   blk, 0, stream>>>(grk, recT);

    for (int step = 0; step < 3; ++step) {
        const float* stf = (step == 0) ? init_state : st;
        float* outst = (step == 2) ? (float*)d_out : st;
        k_edge_transform<<<1024, blk, 0, stream>>>(stf, wT, bvec, hT);
        k_aggregate     <<<512,  blk, 0, stream>>>(wadj, hT, neigh);
        k_gru_gemm      <<<384,  blk, 0, stream>>>(neigh, stf, gkT, recT, gb, tmp);
        k_gru_gates     <<<4096, blk, 0, stream>>>(tmp, stf, zbuf, rhbuf);
        k_gru_update    <<<128,  blk, 0, stream>>>(tmp, zbuf, rhbuf, recT, stf, outst);
    }
}
